// mambalayer_36636071035568
// MI455X (gfx1250) — matrix operand order repaired and checked by reading, not compiled, not run on hardware
//
#include <hip/hip_runtime.h>
#include <hip/hip_bf16.h>
#include <math.h>

// ---------------------------------------------------------------------------
// Problem constants (from reference)
// ---------------------------------------------------------------------------
#define L_SEQ    2048
#define D_MODEL  1024
#define D_INNER  2048
#define D_STATE  16
#define D_CONV   4
#define DT_RANK  64
#define HID      4096
#define LN_EPS   1e-5f

#define SCAN_CHUNK   64
#define SCAN_NCHUNK  (L_SEQ / SCAN_CHUNK)   // 32

typedef __attribute__((ext_vector_type(16))) _Float16 v16h;
typedef __attribute__((ext_vector_type(8)))  _Float16 v8h;
typedef __attribute__((ext_vector_type(8)))  float    v8f;
typedef int v4i __attribute__((vector_size(16)));   // matches builtin prototype

union V16H { v16h v; v8h h[2]; };

// CDNA5 async global->LDS staging (guarded: falls back to load+ds_store)
#if defined(__gfx1250__) && __has_builtin(__builtin_amdgcn_global_load_async_to_lds_b128)
#  define USE_ASYNC_LDS 1
#else
#  define USE_ASYNC_LDS 0
#endif

#define AS1 __attribute__((address_space(1)))
#define AS3 __attribute__((address_space(3)))

__device__ __forceinline__ void wait_stage_async() {
#if USE_ASYNC_LDS
#  if __has_builtin(__builtin_amdgcn_s_wait_asynccnt)
    __builtin_amdgcn_s_wait_asynccnt(0);
#  else
    asm volatile("s_wait_asynccnt 0" ::: "memory");
#  endif
#endif
}

// ---------------------------------------------------------------------------
// f32 -> f16 convert
// ---------------------------------------------------------------------------
__global__ void k_cvt_f16(const float* __restrict__ src, _Float16* __restrict__ dst, int n) {
    int i = blockIdx.x * blockDim.x + threadIdx.x;
    if (i < n) dst[i] = (_Float16)src[i];
}

// ---------------------------------------------------------------------------
// Row LayerNorm: out16 (always), out32 (optional); one block per row
// ---------------------------------------------------------------------------
__global__ void k_ln(const float* __restrict__ x, const float* __restrict__ g,
                     const float* __restrict__ b, _Float16* __restrict__ out16,
                     float* __restrict__ out32, int D) {
    __shared__ float red[256];
    const int row = blockIdx.x;
    const int tid = threadIdx.x;
    const float* xr = x + (size_t)row * D;

    float s = 0.f;
    for (int i = tid; i < D; i += 256) s += xr[i];
    red[tid] = s; __syncthreads();
    for (int o = 128; o > 0; o >>= 1) { if (tid < o) red[tid] += red[tid + o]; __syncthreads(); }
    const float mu = red[0] / (float)D;
    __syncthreads();

    float v = 0.f;
    for (int i = tid; i < D; i += 256) { float d = xr[i] - mu; v += d * d; }
    red[tid] = v; __syncthreads();
    for (int o = 128; o > 0; o >>= 1) { if (tid < o) red[tid] += red[tid + o]; __syncthreads(); }
    const float rstd = rsqrtf(red[0] / (float)D + LN_EPS);

    for (int i = tid; i < D; i += 256) {
        float y = (xr[i] - mu) * rstd * g[i] + b[i];
        out16[(size_t)row * D + i] = (_Float16)y;
        if (out32) out32[(size_t)row * D + i] = y;
    }
}

// ---------------------------------------------------------------------------
// Tiled WMMA GEMM:  C[M,N] = A[M,K] * B[N,K]^T  (TN, both K-major, f16 in/f32 acc)
// Block tile 256x128, 8 waves (4 along M x 2 along N), wave tile 64x64
// (4x4 fragments of 16x16) -> 16 WMMAs per 16 ds_load_b128 per K=32 step.
// Double-buffered LDS; staging uses async global->LDS when available so the
// next tile's DMA overlaps the current tile's WMMAs (ASYNCcnt decoupling).
// EPI 0: Cf = acc
// EPI 1: Cf = softplus(acc + bias[n])
// EPI 2: Ch = (f16) relu(acc + bias[n])
// ---------------------------------------------------------------------------
template <int EPI>
__global__ __launch_bounds__(256)
void gemm_f16_wmma(const _Float16* __restrict__ A, const _Float16* __restrict__ B,
                   const float* __restrict__ bias, float* __restrict__ Cf,
                   _Float16* __restrict__ Ch, int M, int N, int K) {
    constexpr int BM = 256, BN = 128, BK = 32, LDT = BK + 8;
    __shared__ _Float16 As[2][BM][LDT];
    __shared__ _Float16 Bs[2][BN][LDT];

    const int tid  = threadIdx.x;
    const int lane = tid & 31;
    const int wave = tid >> 5;
    const int wm   = wave & 3;   // 4 waves along M, 64 rows each
    const int wn   = wave >> 2;  // 2 waves along N, 64 cols each
    const int block_m = blockIdx.y * BM;
    const int block_n = blockIdx.x * BN;

    v8f acc[4][4];
    const v8f vzero = {};
    #pragma unroll
    for (int i = 0; i < 4; ++i)
        #pragma unroll
        for (int j = 0; j < 4; ++j) acc[i][j] = vzero;

    // cooperative staging indices
    const int ar = tid;                  // A: one full 32-half row per thread
    const int br = tid >> 1;             // B: 128 rows, 16 halves per thread
    const int bc = (tid & 1) * 16;

    const int khalf = lane >> 4;         // lane-half selects K sub-block
    const int mrow  = lane & 15;
    const int ka0 = khalf * 8;           // A elems 0..7  : K 0..7  (lo) / 8..15  (hi)
    const int ka1 = 16 + khalf * 8;      // A elems 8..15 : K 16..23 (lo) / 24..31 (hi)
    const int kb  = khalf * 8;           // B elems 0..7 : K 0..7 (lo) / 8..15 (hi); 8..15 : K 16..23 / 24..31

    auto stage = [&](int p, int k0) {
        const _Float16* Ap = A + (size_t)(block_m + ar) * K + k0;
        _Float16* la = &As[p][ar][0];
        const int brow = block_n + br;
        _Float16* lb = &Bs[p][br][bc];
#if USE_ASYNC_LDS
        __builtin_amdgcn_global_load_async_to_lds_b128((AS1 v4i*)Ap,        (AS3 v4i*)la,        0, 0);
        __builtin_amdgcn_global_load_async_to_lds_b128((AS1 v4i*)(Ap + 8),  (AS3 v4i*)(la + 8),  0, 0);
        __builtin_amdgcn_global_load_async_to_lds_b128((AS1 v4i*)(Ap + 16), (AS3 v4i*)(la + 16), 0, 0);
        __builtin_amdgcn_global_load_async_to_lds_b128((AS1 v4i*)(Ap + 24), (AS3 v4i*)(la + 24), 0, 0);
        if (brow < N) {
            const _Float16* Bp = B + (size_t)brow * K + k0 + bc;
            __builtin_amdgcn_global_load_async_to_lds_b128((AS1 v4i*)Bp,       (AS3 v4i*)lb,       0, 0);
            __builtin_amdgcn_global_load_async_to_lds_b128((AS1 v4i*)(Bp + 8), (AS3 v4i*)(lb + 8), 0, 0);
        } else {
            const v8h z = {};
            *(v8h*)lb = z;
            *(v8h*)(lb + 8) = z;
        }
#else
        *(v8h*)(la)      = *(const v8h*)(Ap);
        *(v8h*)(la + 8)  = *(const v8h*)(Ap + 8);
        *(v8h*)(la + 16) = *(const v8h*)(Ap + 16);
        *(v8h*)(la + 24) = *(const v8h*)(Ap + 24);
        v8h bv0 = {}, bv1 = {};
        if (brow < N) {
            const _Float16* Bp = B + (size_t)brow * K + k0 + bc;
            bv0 = *(const v8h*)Bp;
            bv1 = *(const v8h*)(Bp + 8);
        }
        *(v8h*)lb       = bv0;
        *(v8h*)(lb + 8) = bv1;
        if (k0 + BK < K) {   // prefetch next tile into near caches
            __builtin_prefetch(Ap + BK, 0, 3);
            if (brow < N) __builtin_prefetch(B + (size_t)brow * K + k0 + BK + bc, 0, 3);
        }
#endif
    };

    // prologue: fill buffer 0
    stage(0, 0);
    wait_stage_async();
    __syncthreads();

    int p = 0;
    for (int k0 = 0; k0 < K; k0 += BK) {
        // kick off next tile's staging (overlaps with WMMAs below)
        if (k0 + BK < K) stage(p ^ 1, k0 + BK);

        // ---- fragment loads (contiguous ds_load_b128 pairs) ----
        V16H af[4], bf[4];
        #pragma unroll
        for (int fm = 0; fm < 4; ++fm) {
            const int arow = wm * 64 + fm * 16 + mrow;
            af[fm].h[0] = *(const v8h*)&As[p][arow][ka0];
            af[fm].h[1] = *(const v8h*)&As[p][arow][ka1];
        }
        #pragma unroll
        for (int fn = 0; fn < 4; ++fn) {
            const int bcol = wn * 64 + fn * 16 + mrow;
            bf[fn].h[0] = *(const v8h*)&Bs[p][bcol][kb];
            bf[fn].h[1] = *(const v8h*)&Bs[p][bcol][kb + 16];
        }

        #pragma unroll
        for (int fm = 0; fm < 4; ++fm)
            #pragma unroll
            for (int fn = 0; fn < 4; ++fn)
                acc[fm][fn] = __builtin_amdgcn_wmma_f32_16x16x32_f16(
                    false, af[fm].v, false, bf[fn].v, (short)0, acc[fm][fn], false, false);

        wait_stage_async();   // next tile's async DMA complete (this wave)
        __syncthreads();      // all waves' staging visible
        p ^= 1;
    }

    // ---- epilogue / store (C layout: lane 0-15 -> M 0..7, lane 16-31 -> M 8..15) ----
    #pragma unroll
    for (int fm = 0; fm < 4; ++fm) {
        #pragma unroll
        for (int fn = 0; fn < 4; ++fn) {
            const int m0 = block_m + wm * 64 + fm * 16 + khalf * 8;
            const int n  = block_n + wn * 64 + fn * 16 + mrow;
            if (n >= N) continue;
            #pragma unroll
            for (int r = 0; r < 8; ++r) {
                const int m = m0 + r;
                if (m >= M) continue;
                float v = acc[fm][fn][r];
                if (EPI == 1) {
                    v += bias[n];
                    v = (v > 20.f) ? v : log1pf(__expf(v));      // softplus
                    Cf[(size_t)m * N + n] = v;
                } else if (EPI == 2) {
                    v += bias[n];
                    v = v > 0.f ? v : 0.f;                        // relu
                    Ch[(size_t)m * N + n] = (_Float16)v;
                } else {
                    Cf[(size_t)m * N + n] = v;
                }
            }
        }
    }
}

// ---------------------------------------------------------------------------
// Depthwise causal conv (D_CONV=4) + SiLU over xm = xz[:, :D_INNER]
// ---------------------------------------------------------------------------
__global__ void k_conv_silu(const float* __restrict__ xz, const float* __restrict__ cw,
                            const float* __restrict__ cb, float* __restrict__ xc32,
                            _Float16* __restrict__ xc16) {
    const int idx = blockIdx.x * blockDim.x + threadIdx.x;
    if (idx >= L_SEQ * D_INNER) return;
    const int l = idx / D_INNER;
    const int c = idx % D_INNER;
    float acc = cb[c];
    #pragma unroll
    for (int k = 0; k < D_CONV; ++k) {
        const int ls = l - (D_CONV - 1) + k;
        if (ls >= 0) acc += xz[(size_t)ls * (2 * D_INNER) + c] * cw[c * D_CONV + k];
    }
    acc = acc / (1.f + __expf(-acc));   // silu
    xc32[(size_t)l * D_INNER + c] = acc;
    xc16[(size_t)l * D_INNER + c] = (_Float16)acc;
}

// dt16[l, r] = (f16) proj[l, r]
__global__ void k_dt_extract(const float* __restrict__ proj, _Float16* __restrict__ dt16) {
    const int idx = blockIdx.x * blockDim.x + threadIdx.x;
    if (idx >= L_SEQ * DT_RANK) return;
    const int l = idx / DT_RANK;
    const int r = idx % DT_RANK;
    dt16[(size_t)l * DT_RANK + r] = (_Float16)proj[(size_t)l * (DT_RANK + 2 * D_STATE) + r];
}

// ---------------------------------------------------------------------------
// Chunked parallel selective scan (3 passes).
// Per (channel, state) the recurrence is linear:  h <- exp(d*A)h + d*B*x,
// and the chunk transition factor is exp(A * sum(d)) -> cheap to compose.
// ---------------------------------------------------------------------------

// Pass 1: local scan per chunk with h0 = 0; record final h and sum(delta).
__global__ __launch_bounds__(256)
void k_scan_pass1(const float* __restrict__ delta, const float* __restrict__ proj,
                  const float* __restrict__ xc32, const float* __restrict__ A_log,
                  float* __restrict__ hloc, float* __restrict__ sumd) {
    const int tid = threadIdx.x;
    const int c   = blockIdx.x * blockDim.x + tid;
    const int k   = blockIdx.y;
    const int l0  = k * SCAN_CHUNK;
    __shared__ float sB[D_STATE];

    float Aneg[D_STATE], h[D_STATE];
    #pragma unroll
    for (int s = 0; s < D_STATE; ++s) {
        Aneg[s] = -__expf(A_log[(size_t)c * D_STATE + s]);
        h[s] = 0.f;
    }
    float sd = 0.f;

    for (int l = l0; l < l0 + SCAN_CHUNK; ++l) {
        if (tid < D_STATE)
            sB[tid] = proj[(size_t)l * (DT_RANK + 2 * D_STATE) + DT_RANK + tid];
        __syncthreads();
        const float d  = delta[(size_t)l * D_INNER + c];
        const float xv = xc32[(size_t)l * D_INNER + c];
        sd += d;
        #pragma unroll
        for (int s = 0; s < D_STATE; ++s) {
            const float dA = __expf(d * Aneg[s]);
            h[s] = h[s] * dA + d * sB[s] * xv;
        }
        __syncthreads();
    }
    #pragma unroll
    for (int s = 0; s < D_STATE; ++s)
        hloc[((size_t)k * D_INNER + c) * D_STATE + s] = h[s];
    sumd[(size_t)k * D_INNER + c] = sd;
}

// Pass 2: tiny sequential pass over chunks -> true h0 per chunk.
__global__ __launch_bounds__(256)
void k_scan_pass2(const float* __restrict__ A_log, const float* __restrict__ sumd,
                  const float* __restrict__ hloc, float* __restrict__ hstart) {
    const int c = blockIdx.x * blockDim.x + threadIdx.x;
    float Aneg[D_STATE], h[D_STATE];
    #pragma unroll
    for (int s = 0; s < D_STATE; ++s) {
        Aneg[s] = -__expf(A_log[(size_t)c * D_STATE + s]);
        h[s] = 0.f;
    }
    for (int k = 0; k < SCAN_NCHUNK; ++k) {
        const size_t base = ((size_t)k * D_INNER + c) * D_STATE;
        const float sd = sumd[(size_t)k * D_INNER + c];
        #pragma unroll
        for (int s = 0; s < D_STATE; ++s) {
            hstart[base + s] = h[s];
            const float P = __expf(Aneg[s] * sd);     // product of dA over chunk
            h[s] = h[s] * P + hloc[base + s];
        }
    }
}

// Pass 3: local re-scan with true h0; emit gated output y16.
__global__ __launch_bounds__(256)
void k_scan_pass3(const float* __restrict__ delta, const float* __restrict__ proj,
                  const float* __restrict__ xc32, const float* __restrict__ xz,
                  const float* __restrict__ A_log, const float* __restrict__ Dp,
                  const float* __restrict__ hstart, _Float16* __restrict__ y16) {
    const int tid = threadIdx.x;
    const int c   = blockIdx.x * blockDim.x + tid;
    const int k   = blockIdx.y;
    const int l0  = k * SCAN_CHUNK;
    __shared__ float sBC[2 * D_STATE];   // [0..15]=B, [16..31]=C

    float Aneg[D_STATE], h[D_STATE];
    const size_t base = ((size_t)k * D_INNER + c) * D_STATE;
    #pragma unroll
    for (int s = 0; s < D_STATE; ++s) {
        Aneg[s] = -__expf(A_log[(size_t)c * D_STATE + s]);
        h[s] = hstart[base + s];
    }
    const float dpar = Dp[c];

    for (int l = l0; l < l0 + SCAN_CHUNK; ++l) {
        if (tid < 2 * D_STATE)
            sBC[tid] = proj[(size_t)l * (DT_RANK + 2 * D_STATE) + DT_RANK + tid];
        __syncthreads();
        const float d  = delta[(size_t)l * D_INNER + c];
        const float xv = xc32[(size_t)l * D_INNER + c];
        float acc = 0.f;
        #pragma unroll
        for (int s = 0; s < D_STATE; ++s) {
            const float dA = __expf(d * Aneg[s]);
            h[s] = h[s] * dA + d * sBC[s] * xv;
            acc += h[s] * sBC[D_STATE + s];
        }
        float y = acc + xv * dpar;
        const float z = xz[(size_t)l * (2 * D_INNER) + D_INNER + c];
        y *= z / (1.f + __expf(-z));     // y * silu(z)
        y16[(size_t)l * D_INNER + c] = (_Float16)y;
        __syncthreads();
    }
}

// out = a + b (elementwise)
__global__ void k_add(const float* __restrict__ a, const float* __restrict__ b,
                      float* __restrict__ out, int n) {
    int i = blockIdx.x * blockDim.x + threadIdx.x;
    if (i < n) out[i] = a[i] + b[i];
}

// out = h + n + ffn_raw + b2[col]
__global__ void k_final(const float* __restrict__ h, const float* __restrict__ nrm,
                        const float* __restrict__ raw, const float* __restrict__ b2,
                        float* __restrict__ out) {
    int i = blockIdx.x * blockDim.x + threadIdx.x;
    if (i >= L_SEQ * D_MODEL) return;
    const int d = i % D_MODEL;
    out[i] = h[i] + nrm[i] + raw[i] + b2[d];
}

// ---------------------------------------------------------------------------
// Launch
// ---------------------------------------------------------------------------
extern "C" void kernel_launch(void* const* d_in, const int* in_sizes, int n_in,
                              void* d_out, int out_size, void* d_ws, size_t ws_size,
                              hipStream_t stream) {
    (void)in_sizes; (void)n_in; (void)out_size; (void)ws_size;

    const float* x        = (const float*)d_in[0];
    const float* ln1_g    = (const float*)d_in[1];
    const float* ln1_b    = (const float*)d_in[2];
    const float* ln2_g    = (const float*)d_in[3];
    const float* ln2_b    = (const float*)d_in[4];
    const float* in_w     = (const float*)d_in[5];   // (4096, 1024)
    const float* conv_w   = (const float*)d_in[6];   // (2048, 4)
    const float* conv_b   = (const float*)d_in[7];
    const float* xproj_w  = (const float*)d_in[8];   // (96, 2048)
    const float* dtproj_w = (const float*)d_in[9];   // (2048, 64)
    const float* dtproj_b = (const float*)d_in[10];
    const float* A_log    = (const float*)d_in[11];
    const float* D_param  = (const float*)d_in[12];
    const float* out_w    = (const float*)d_in[13];  // (1024, 2048)
    const float* ffn_w1   = (const float*)d_in[14];  // (4096, 1024)
    const float* ffn_b1   = (const float*)d_in[15];
    const float* ffn_w2   = (const float*)d_in[16];  // (1024, 4096)
    const float* ffn_b2   = (const float*)d_in[17];
    float* out = (float*)d_out;

    // -------- workspace arena --------
    char* ws = (char*)d_ws;
    size_t off = 0;
    auto alloc = [&](size_t bytes) { size_t r = off; off = (off + bytes + 255) & ~(size_t)255; return r; };

    _Float16* Win16   = (_Float16*)(ws + alloc((size_t)2 * D_INNER * D_MODEL * 2));
    _Float16* Wxp16   = (_Float16*)(ws + alloc((size_t)(DT_RANK + 2 * D_STATE) * D_INNER * 2));
    _Float16* Wdt16   = (_Float16*)(ws + alloc((size_t)D_INNER * DT_RANK * 2));
    _Float16* Wout16  = (_Float16*)(ws + alloc((size_t)D_MODEL * D_INNER * 2));
    _Float16* W1_16   = (_Float16*)(ws + alloc((size_t)HID * D_MODEL * 2));
    _Float16* W2_16   = (_Float16*)(ws + alloc((size_t)D_MODEL * HID * 2));
    _Float16* hnorm16 = (_Float16*)(ws + alloc((size_t)L_SEQ * D_MODEL * 2));
    float*    xz32    = (float*)   (ws + alloc((size_t)L_SEQ * 2 * D_INNER * 4));
    float*    xc32    = (float*)   (ws + alloc((size_t)L_SEQ * D_INNER * 4));
    _Float16* xc16    = (_Float16*)(ws + alloc((size_t)L_SEQ * D_INNER * 2));
    float*    proj32  = (float*)   (ws + alloc((size_t)L_SEQ * (DT_RANK + 2 * D_STATE) * 4));
    _Float16* dt16    = (_Float16*)(ws + alloc((size_t)L_SEQ * DT_RANK * 2));
    float*    delta32 = (float*)   (ws + alloc((size_t)L_SEQ * D_INNER * 4));
    _Float16* y16     = (_Float16*)(ws + alloc((size_t)L_SEQ * D_INNER * 2));
    float*    mraw32  = (float*)   (ws + alloc((size_t)L_SEQ * D_MODEL * 4));
    float*    h32     = (float*)   (ws + alloc((size_t)L_SEQ * D_MODEL * 4));
    _Float16* n16     = (_Float16*)(ws + alloc((size_t)L_SEQ * D_MODEL * 2));
    float*    n32     = (float*)   (ws + alloc((size_t)L_SEQ * D_MODEL * 4));
    _Float16* mid16   = (_Float16*)(ws + alloc((size_t)L_SEQ * HID * 2));
    float*    fraw32  = (float*)   (ws + alloc((size_t)L_SEQ * D_MODEL * 4));
    float*    hloc    = (float*)   (ws + alloc((size_t)SCAN_NCHUNK * D_INNER * D_STATE * 4));
    float*    hstart  = (float*)   (ws + alloc((size_t)SCAN_NCHUNK * D_INNER * D_STATE * 4));
    float*    sumd    = (float*)   (ws + alloc((size_t)SCAN_NCHUNK * D_INNER * 4));

    auto cvt = [&](const float* s, _Float16* d, size_t n) {
        k_cvt_f16<<<dim3((unsigned)((n + 255) / 256)), dim3(256), 0, stream>>>(s, d, (int)n);
    };
    auto gemm_grid = [](int M, int N) { return dim3((unsigned)((N + 127) / 128), (unsigned)(M / 256)); };

    // 0) weight conversions f32 -> f16
    cvt(in_w,     Win16,  (size_t)2 * D_INNER * D_MODEL);
    cvt(xproj_w,  Wxp16,  (size_t)(DT_RANK + 2 * D_STATE) * D_INNER);
    cvt(dtproj_w, Wdt16,  (size_t)D_INNER * DT_RANK);
    cvt(out_w,    Wout16, (size_t)D_MODEL * D_INNER);
    cvt(ffn_w1,   W1_16,  (size_t)HID * D_MODEL);
    cvt(ffn_w2,   W2_16,  (size_t)D_MODEL * HID);

    // 1) LN1 -> f16
    k_ln<<<dim3(L_SEQ), dim3(256), 0, stream>>>(x, ln1_g, ln1_b, hnorm16, nullptr, D_MODEL);

    // 2) xz = hnorm @ in_w^T   (2048 x 4096, K=1024)
    gemm_f16_wmma<0><<<gemm_grid(L_SEQ, 2 * D_INNER), dim3(256), 0, stream>>>(
        hnorm16, Win16, nullptr, xz32, nullptr, L_SEQ, 2 * D_INNER, D_MODEL);

    // 3) depthwise conv + silu
    k_conv_silu<<<dim3((L_SEQ * D_INNER + 255) / 256), dim3(256), 0, stream>>>(
        xz32, conv_w, conv_b, xc32, xc16);

    // 4) proj = xc @ xproj_w^T   (2048 x 96, K=2048)
    gemm_f16_wmma<0><<<gemm_grid(L_SEQ, DT_RANK + 2 * D_STATE), dim3(256), 0, stream>>>(
        xc16, Wxp16, nullptr, proj32, nullptr, L_SEQ, DT_RANK + 2 * D_STATE, D_INNER);

    // 5) dt slice -> f16
    k_dt_extract<<<dim3((L_SEQ * DT_RANK + 255) / 256), dim3(256), 0, stream>>>(proj32, dt16);

    // 6) delta = softplus(dt @ dtproj_w^T + b)   (2048 x 2048, K=64)
    gemm_f16_wmma<1><<<gemm_grid(L_SEQ, D_INNER), dim3(256), 0, stream>>>(
        dt16, Wdt16, dtproj_b, delta32, nullptr, L_SEQ, D_INNER, DT_RANK);

    // 7) chunked parallel selective scan -> y16
    k_scan_pass1<<<dim3(D_INNER / 256, SCAN_NCHUNK), dim3(256), 0, stream>>>(
        delta32, proj32, xc32, A_log, hloc, sumd);
    k_scan_pass2<<<dim3(D_INNER / 256), dim3(256), 0, stream>>>(
        A_log, sumd, hloc, hstart);
    k_scan_pass3<<<dim3(D_INNER / 256, SCAN_NCHUNK), dim3(256), 0, stream>>>(
        delta32, proj32, xc32, xz32, A_log, D_param, hstart, y16);

    // 8) mamba_raw = y @ out_w^T   (2048 x 1024, K=2048)
    gemm_f16_wmma<0><<<gemm_grid(L_SEQ, D_MODEL), dim3(256), 0, stream>>>(
        y16, Wout16, nullptr, mraw32, nullptr, L_SEQ, D_MODEL, D_INNER);

    // 9) h = mamba + residual
    k_add<<<dim3((L_SEQ * D_MODEL + 255) / 256), dim3(256), 0, stream>>>(
        mraw32, x, h32, L_SEQ * D_MODEL);

    // 10) LN2 -> n16 + n32
    k_ln<<<dim3(L_SEQ), dim3(256), 0, stream>>>(h32, ln2_g, ln2_b, n16, n32, D_MODEL);

    // 11) mid = relu(n @ w1^T + b1) -> f16   (2048 x 4096, K=1024)
    gemm_f16_wmma<2><<<gemm_grid(L_SEQ, HID), dim3(256), 0, stream>>>(
        n16, W1_16, ffn_b1, nullptr, mid16, L_SEQ, HID, D_MODEL);

    // 12) fraw = mid @ w2^T   (2048 x 1024, K=4096)
    gemm_f16_wmma<0><<<gemm_grid(L_SEQ, D_MODEL), dim3(256), 0, stream>>>(
        mid16, W2_16, nullptr, fraw32, nullptr, L_SEQ, D_MODEL, HID);

    // 13) out = h + n + fraw + b2
    k_final<<<dim3((L_SEQ * D_MODEL + 255) / 256), dim3(256), 0, stream>>>(
        h32, n32, fraw32, ffn_b2, out);
}
